// MLPPredictor2_11269994185515
// MI455X (gfx1250) — matrix operand order repaired and checked by reading, not compiled, not run on hardware
//
#include <hip/hip_runtime.h>

typedef __attribute__((ext_vector_type(16))) _Float16 v16h;
typedef __attribute__((ext_vector_type(8)))  _Float16 v8h;
typedef __attribute__((ext_vector_type(8)))  float    v8f;

#define NSAMP  250000
#define NNODES 100000
#define DFEAT  128
#define NHID   64
#define NELEM  (NNODES * DFEAT)   // 12,800,000 per feature table

// ---------------------------------------------------------------------------
// Pre-pass: convert both feature tables f32 -> f16 (vectorized, 8 elem/thread)
// ---------------------------------------------------------------------------
__global__ __launch_bounds__(256) void cvt_f32_to_f16(
    const float* __restrict__ a, const float* __restrict__ b,
    _Float16* __restrict__ oa, _Float16* __restrict__ ob)
{
    const long long i = ((long long)blockIdx.x * 256 + threadIdx.x) * 8;
    if (i >= NELEM) return;
    float4 x0 = *(const float4*)(a + i);
    float4 x1 = *(const float4*)(a + i + 4);
    float4 y0 = *(const float4*)(b + i);
    float4 y1 = *(const float4*)(b + i + 4);
    v8h ha, hb;
    ha[0] = (_Float16)x0.x; ha[1] = (_Float16)x0.y;
    ha[2] = (_Float16)x0.z; ha[3] = (_Float16)x0.w;
    ha[4] = (_Float16)x1.x; ha[5] = (_Float16)x1.y;
    ha[6] = (_Float16)x1.z; ha[7] = (_Float16)x1.w;
    hb[0] = (_Float16)y0.x; hb[1] = (_Float16)y0.y;
    hb[2] = (_Float16)y0.z; hb[3] = (_Float16)y0.w;
    hb[4] = (_Float16)y1.x; hb[5] = (_Float16)y1.y;
    hb[6] = (_Float16)y1.z; hb[7] = (_Float16)y1.w;
    *(v8h*)(oa + i) = ha;
    *(v8h*)(ob + i) = hb;
}

// ---------------------------------------------------------------------------
// Shared: fill LDS with w1 pre-swizzled into WMMA B-fragment layout (f16).
// Fragment f = kc*4+nt holds the 32(K)x16(N) tile; per lane 16 halves,
// half j <-> K = kc*32 + (lane>=16 ? 8 : 0) + (j&7) + (j>=8 ? 16 : 0), N = nt*16 + (lane&15).
// ---------------------------------------------------------------------------
__device__ __forceinline__ void fill_ldsB(_Float16* ldsB, const float* __restrict__ w1,
                                          int tid)
{
    for (int idx = tid; idx < 16 * 4 * 512; idx += 256) {
        int f  = idx >> 9;
        int r  = idx & 511;
        int ln = r >> 4;
        int j  = r & 15;
        int kc = f >> 2, nt = f & 3;
        int n  = nt * 16 + (ln & 15);
        int k  = kc * 32 + ((ln >> 4) << 3) + (j & 7) + ((j >> 3) << 4);
        ldsB[idx] = (_Float16)w1[k * NHID + n];
    }
}

// ---------------------------------------------------------------------------
// Main kernel, f16 feature path: each wave computes a 64(M) x 64(N) tile.
// 4 A tiles reuse every B fragment -> 4 WMMAs per ds_load_b128 pair.
// ---------------------------------------------------------------------------
__global__ __launch_bounds__(256) void mlp_link_f16(
    const _Float16* __restrict__ f1, const _Float16* __restrict__ f2,
    const long long* __restrict__ ts, const float* __restrict__ w1,
    const float* __restrict__ b1, const float* __restrict__ w2,
    float* __restrict__ out)
{
    __shared__ __attribute__((aligned(32))) _Float16 ldsB[16 * 4 * 512]; // 64 KB

    const int tid = threadIdx.x;
    fill_ldsB(ldsB, w1, tid);
    __syncthreads();

    const int wave = tid >> 5;
    const int lane = tid & 31;
    const int lm   = lane & 15;
    const int hi   = lane >> 4;

    const long long wave_m = ((long long)blockIdx.x * 8 + wave) * 64;
    if (wave_m >= NSAMP) return;   // uniform per wave

    // Per-tile gather offsets (bytes into the f16 tables); clamp tail tiles so
    // EXEC stays all-ones for WMMA, guard their stores instead.
    unsigned osrc[4], odst[4];
    bool valid[4];
    #pragma unroll
    for (int t = 0; t < 4; ++t) {
        long long m = wave_m + t * 16 + lm;
        valid[t] = (wave_m + t * 16) < NSAMP;   // NSAMP % 16 == 0
        if (m >= NSAMP) m = NSAMP - 1;
        osrc[t] = (unsigned)ts[2 * m]     * (DFEAT * 2);
        odst[t] = (unsigned)ts[2 * m + 1] * (DFEAT * 2);
    }
    const char* f1c = (const char*)f1;
    const char* f2c = (const char*)f2;

    v8f acc[4][4] = {};

    #pragma unroll
    for (int kc = 0; kc < 16; ++kc) {
        // seg: which 128-col segment of h this K-chunk lives in.
        const int seg  = kc >> 2;
        const char* base = (seg < 2) ? f1c : f2c;
        const bool  use_dst = (seg & 1);
        // Byte offset of this lane's first 8 halves within the row:
        // column = (kc&3)*32 + hi*8  ->  bytes = (kc&3)*64 + hi*16.
        const int c0 = ((kc & 3) << 6) + hi * 16;

        v16h a[4];
        #pragma unroll
        for (int t = 0; t < 4; ++t) {
            const char* row = base + (use_dst ? odst[t] : osrc[t]);
            v8h lo  = *(const v8h*)(row + c0);        // halves 0..7  (K = hi*8+0..7)
            v8h hi8 = *(const v8h*)(row + c0 + 32);   // halves 8..15 (K = 16+hi*8+0..7)
            a[t] = __builtin_shufflevector(lo, hi8,
                                           0, 1, 2, 3, 4, 5, 6, 7,
                                           8, 9, 10, 11, 12, 13, 14, 15);
        }

        #pragma unroll
        for (int nt = 0; nt < 4; ++nt) {
            const v16h b = *(const v16h*)&ldsB[(kc * 4 + nt) * 512 + lane * 16];
            #pragma unroll
            for (int t = 0; t < 4; ++t) {
                acc[t][nt] = __builtin_amdgcn_wmma_f32_16x16x32_f16(
                    false, a[t], false, b, (short)0, acc[t][nt], false, false);
            }
        }
    }

    // Epilogue: relu(acc + b1) . w2 reduced over N=64.
    // C/D layout: lane group hi holds rows M = hi*8 + r, column N = nt*16 + lm.
    #pragma unroll
    for (int t = 0; t < 4; ++t) {
        float part[8] = {0, 0, 0, 0, 0, 0, 0, 0};
        #pragma unroll
        for (int nt = 0; nt < 4; ++nt) {
            const int n    = nt * 16 + lm;
            const float bn = b1[n];
            const float wn = w2[n];
            #pragma unroll
            for (int r = 0; r < 8; ++r) {
                float v = acc[t][nt][r] + bn;
                v = v > 0.0f ? v : 0.0f;
                part[r] += v * wn;
            }
        }
        #pragma unroll
        for (int off = 1; off <= 8; off <<= 1) {
            #pragma unroll
            for (int r = 0; r < 8; ++r)
                part[r] += __shfl_xor(part[r], off, 32);
        }
        if (lm == 0 && valid[t]) {
            #pragma unroll
            for (int r = 0; r < 8; ++r)
                out[wave_m + t * 16 + hi * 8 + r] = part[r];
        }
    }
}

// ---------------------------------------------------------------------------
// Fallback (f32 gather, 32 M/wave) -- used only if d_ws is too small for the
// converted feature tables.
// ---------------------------------------------------------------------------
__global__ __launch_bounds__(256) void mlp_link_f32(
    const float* __restrict__ in1, const float* __restrict__ in2,
    const long long* __restrict__ ts, const float* __restrict__ w1,
    const float* __restrict__ b1, const float* __restrict__ w2,
    float* __restrict__ out)
{
    __shared__ __attribute__((aligned(32))) _Float16 ldsB[16 * 4 * 512];

    const int tid = threadIdx.x;
    fill_ldsB(ldsB, w1, tid);
    __syncthreads();

    const int wave = tid >> 5;
    const int lane = tid & 31;
    const int lm   = lane & 15;
    const int hi   = lane >> 4;

    const long long wave_m = ((long long)blockIdx.x * 8 + wave) * 32;
    if (wave_m >= NSAMP) return;

    const long long m1 = wave_m + lm;
    long long m2 = wave_m + 16 + lm;
    const bool tile2_valid = (wave_m + 16) < NSAMP;
    if (m2 >= NSAMP) m2 = NSAMP - 1;

    const long long s1 = ts[2 * m1], d1 = ts[2 * m1 + 1];
    const long long s2 = ts[2 * m2], d2 = ts[2 * m2 + 1];
    const float* rowp1[4];
    const float* rowp2[4];
    rowp1[0] = in1 + s1 * DFEAT;  rowp2[0] = in1 + s2 * DFEAT;
    rowp1[1] = in1 + d1 * DFEAT;  rowp2[1] = in1 + d2 * DFEAT;
    rowp1[2] = in2 + s1 * DFEAT;  rowp2[2] = in2 + s2 * DFEAT;
    rowp1[3] = in2 + d1 * DFEAT;  rowp2[3] = in2 + d2 * DFEAT;

    v8f acc1[4] = {};
    v8f acc2[4] = {};

    #pragma unroll
    for (int kc = 0; kc < 16; ++kc) {
        const int coff = ((kc & 3) << 5) + hi * 8;
        const float* p1 = rowp1[kc >> 2] + coff;
        const float* p2 = rowp2[kc >> 2] + coff;

        float4 x0 = *(const float4*)(p1);
        float4 x1 = *(const float4*)(p1 + 4);
        float4 x2 = *(const float4*)(p1 + 16);
        float4 x3 = *(const float4*)(p1 + 20);
        float4 y0 = *(const float4*)(p2);
        float4 y1 = *(const float4*)(p2 + 4);
        float4 y2 = *(const float4*)(p2 + 16);
        float4 y3 = *(const float4*)(p2 + 20);

        v16h a1, a2;
        a1[0]  = (_Float16)x0.x; a1[1]  = (_Float16)x0.y;
        a1[2]  = (_Float16)x0.z; a1[3]  = (_Float16)x0.w;
        a1[4]  = (_Float16)x1.x; a1[5]  = (_Float16)x1.y;
        a1[6]  = (_Float16)x1.z; a1[7]  = (_Float16)x1.w;
        a1[8]  = (_Float16)x2.x; a1[9]  = (_Float16)x2.y;
        a1[10] = (_Float16)x2.z; a1[11] = (_Float16)x2.w;
        a1[12] = (_Float16)x3.x; a1[13] = (_Float16)x3.y;
        a1[14] = (_Float16)x3.z; a1[15] = (_Float16)x3.w;
        a2[0]  = (_Float16)y0.x; a2[1]  = (_Float16)y0.y;
        a2[2]  = (_Float16)y0.z; a2[3]  = (_Float16)y0.w;
        a2[4]  = (_Float16)y1.x; a2[5]  = (_Float16)y1.y;
        a2[6]  = (_Float16)y1.z; a2[7]  = (_Float16)y1.w;
        a2[8]  = (_Float16)y2.x; a2[9]  = (_Float16)y2.y;
        a2[10] = (_Float16)y2.z; a2[11] = (_Float16)y2.w;
        a2[12] = (_Float16)y3.x; a2[13] = (_Float16)y3.y;
        a2[14] = (_Float16)y3.z; a2[15] = (_Float16)y3.w;

        #pragma unroll
        for (int nt = 0; nt < 4; ++nt) {
            const v16h b = *(const v16h*)&ldsB[(kc * 4 + nt) * 512 + lane * 16];
            acc1[nt] = __builtin_amdgcn_wmma_f32_16x16x32_f16(
                false, a1, false, b, (short)0, acc1[nt], false, false);
            acc2[nt] = __builtin_amdgcn_wmma_f32_16x16x32_f16(
                false, a2, false, b, (short)0, acc2[nt], false, false);
        }
    }

    float p1s[8] = {0, 0, 0, 0, 0, 0, 0, 0};
    float p2s[8] = {0, 0, 0, 0, 0, 0, 0, 0};
    #pragma unroll
    for (int nt = 0; nt < 4; ++nt) {
        const int n    = nt * 16 + lm;
        const float bn = b1[n];
        const float wn = w2[n];
        #pragma unroll
        for (int r = 0; r < 8; ++r) {
            float v1 = acc1[nt][r] + bn;
            v1 = v1 > 0.0f ? v1 : 0.0f;
            p1s[r] += v1 * wn;
            float v2 = acc2[nt][r] + bn;
            v2 = v2 > 0.0f ? v2 : 0.0f;
            p2s[r] += v2 * wn;
        }
    }
    #pragma unroll
    for (int off = 1; off <= 8; off <<= 1) {
        #pragma unroll
        for (int r = 0; r < 8; ++r) {
            p1s[r] += __shfl_xor(p1s[r], off, 32);
            p2s[r] += __shfl_xor(p2s[r], off, 32);
        }
    }
    if (lm == 0) {
        #pragma unroll
        for (int r = 0; r < 8; ++r)
            out[wave_m + hi * 8 + r] = p1s[r];
        if (tile2_valid) {
            #pragma unroll
            for (int r = 0; r < 8; ++r)
                out[wave_m + 16 + hi * 8 + r] = p2s[r];
        }
    }
}

extern "C" void kernel_launch(void* const* d_in, const int* in_sizes, int n_in,
                              void* d_out, int out_size, void* d_ws, size_t ws_size,
                              hipStream_t stream) {
    const float*     in1 = (const float*)d_in[0];
    const float*     in2 = (const float*)d_in[1];
    const long long* ts  = (const long long*)d_in[2];
    const float*     w1  = (const float*)d_in[3];
    const float*     b1  = (const float*)d_in[4];
    const float*     w2  = (const float*)d_in[5];
    float* out = (float*)d_out;

    const size_t need = (size_t)2 * NELEM * sizeof(_Float16);   // 51.2 MB
    if (ws_size >= need) {
        _Float16* f1 = (_Float16*)d_ws;
        _Float16* f2 = f1 + NELEM;
        cvt_f32_to_f16<<<NELEM / 8 / 256, 256, 0, stream>>>(in1, in2, f1, f2);
        // 8 waves/block * 64 samples/wave = 512 samples/block
        mlp_link_f16<<<(NSAMP + 511) / 512, 256, 0, stream>>>(f1, f2, ts, w1, b1, w2, out);
    } else {
        mlp_link_f32<<<(NSAMP + 255) / 256, 256, 0, stream>>>(in1, in2, ts, w1, b1, w2, out);
    }
}
